// Attention_81217831567599
// MI455X (gfx1250) — matrix operand order repaired and checked by reading, not compiled, not run on hardware
//
#include <hip/hip_runtime.h>

typedef __attribute__((ext_vector_type(16))) __bf16 v16bf;
typedef __attribute__((ext_vector_type(8)))  __bf16 v8bf;
typedef __attribute__((ext_vector_type(4)))  __bf16 v4bf;
typedef __attribute__((ext_vector_type(8)))  float  v8f;
typedef int v4i_t __attribute__((ext_vector_type(4)));
typedef int v8i_t __attribute__((ext_vector_type(8)));

#define BB 2
#define TT 2048
#define CC 1024
#define HH 16
#define DD 64

__device__ __forceinline__ v16bf cat8(v8bf lo, v8bf hi) {
  return __builtin_shufflevector(lo, hi, 0,1,2,3,4,5,6,7,8,9,10,11,12,13,14,15);
}

__device__ __forceinline__ v8f wmma_bf16(v16bf a, v16bf b, v8f c) {
  return __builtin_amdgcn_wmma_f32_16x16x32_bf16(false, a, false, b, (short)0, c,
                                                 false, false);
}

// A-fragment (16x32 bf16): lane holds row (lane%16); k = 16*(lane/16)+e.
__device__ __forceinline__ v16bf ldfragA(const __bf16* base, int row, int lhi) {
  const __bf16* p = base + row * 40 + 16 * lhi;
  return cat8(*(const v8bf*)p, *(const v8bf*)(p + 8));
}
// B-fragment (32x16 bf16, tile stored [n][k]): lane holds col (lane%16); k = 16*(lane/16)+e.
__device__ __forceinline__ v16bf ldfragB(const __bf16* base, int row, int lhi) {
  const __bf16* p = base + row * 40 + 16 * lhi;
  return cat8(*(const v8bf*)p, *(const v8bf*)(p + 8));
}

// Async-stage a ROWSx32 bf16 tile (row-major, source row stride CC elems) into LDS
// with 40-element padded rows. 16B per lane per issue, ASYNCcnt-tracked, no VGPR data.
template <int ROWS>
__device__ __forceinline__ void stage_tile_async(const __bf16* __restrict__ g,
                                                 unsigned ldsbase, int m0, int k0,
                                                 int tid) {
#pragma unroll
  for (int i = 0; i < ROWS / 64; ++i) {
    int c = i * 256 + tid;                  // chunk id; 4 x 16B chunks per row
    int row = c >> 2, co = (c & 3) << 3;    // co = element offset in row
    unsigned lds = ldsbase + (unsigned)((row * 40 + co) * 2);
    unsigned goff = (unsigned)(((m0 + row) * CC + k0 + co) * 2);
    asm volatile("global_load_async_to_lds_b128 %0, %1, %2"
                 :: "v"(lds), "v"(goff),
                    "s"(reinterpret_cast<unsigned long long>(g))
                 : "memory");
  }
}

// TDM: DMA a 128x32 bf16 tile of a [CC x CC] bf16 tensor (tile origin n0,k0) into LDS,
// padding each 64B row to 80B (40-element stride). TENSORcnt-tracked; wave-level issue.
__device__ __forceinline__ void tdm_load_w_tile(const __bf16* __restrict__ wb,
                                                unsigned ldsoff, int n0, int k0) {
  unsigned long long ga =
      reinterpret_cast<unsigned long long>(wb + ((size_t)n0 * CC + k0));
  v4i_t g0;
  g0[0] = 1;                                            // count=1, user mode
  g0[1] = (int)ldsoff;                                  // lds_addr
  g0[2] = (int)(unsigned)(ga & 0xffffffffu);            // global_addr[31:0]
  g0[3] = (int)((unsigned)((ga >> 32) & 0x01ffffffu)    // global_addr[56:32]
                | 0x80000000u);                         // type=2 ("image")
  v8i_t g1;
  g1[0] = (1 << 16)      // data_size = 2 bytes
        | (1 << 20)      // pad_enable
        | (3 << 22)      // pad_interval: 16 DWORDs (=64B row)
        | (3 << 25);     // pad_amount:   4 DWORDs (=16B pad -> 80B stride)
  g1[1] = (int)(((unsigned)CC & 0xffffu) << 16);        // tensor_dim0[15:0]
  g1[2] = (int)(((unsigned)CC >> 16)                    // tensor_dim0[31:16]
                | (((unsigned)CC & 0xffffu) << 16));    // tensor_dim1[15:0]
  g1[3] = (int)(((unsigned)CC >> 16) | (32u << 16));    // tensor_dim1[31:16] | tile_dim0=32
  g1[4] = 128;                                          // tile_dim1=128, tile_dim2=0
  g1[5] = CC;                                           // tensor_dim0_stride[31:0]
  g1[6] = 0;                                            // stride hi | dim1_stride lo
  g1[7] = 0;
  asm volatile("tensor_load_to_lds %0, %1" :: "s"(g0), "s"(g1) : "memory");
}

// ---------------------------------------------------------------------------
// Kernel 0: bulk f32 -> bf16 convert (hoists conversion out of GEMM hot loops)
// ---------------------------------------------------------------------------
__global__ __launch_bounds__(256)
void cvt_f32_bf16(const float* __restrict__ src, __bf16* __restrict__ dst, int n4) {
  int i = blockIdx.x * 256 + threadIdx.x;
  if (i < n4) {
    float4 f = *(const float4*)(src + (size_t)i * 4);
    v4bf o; o[0] = (__bf16)f.x; o[1] = (__bf16)f.y; o[2] = (__bf16)f.z; o[3] = (__bf16)f.w;
    *(v4bf*)(dst + (size_t)i * 4) = o;
  }
}

// ---------------------------------------------------------------------------
// Kernel 1: qkv = x @ w_qkv^T  (M=4096, N=3072, K=1024), async double-buffered,
// fused RoPE; q/k -> [sec][B][H][T][D] bf16, v -> [B][H][D][T] bf16 (transposed)
// ---------------------------------------------------------------------------
__global__ __launch_bounds__(256)
void qkv_rope_gemm(const __bf16* __restrict__ xb, const __bf16* __restrict__ wb,
                   __bf16* __restrict__ qk, __bf16* __restrict__ vt) {
  __shared__ __bf16 As[2][64 * 40];
  __shared__ __bf16 Bs[2][128 * 40];
  const int tid = threadIdx.x;
  const int lane = tid & 31, wid = tid >> 5;
  const int lrow = lane & 15, lhi = lane >> 4;
  const int wm = wid >> 2, wn = wid & 3;
  const int m0 = blockIdx.y * 64;
  const int n0 = blockIdx.x * 128;
  const unsigned as0 = (unsigned)reinterpret_cast<unsigned long long>(&As[0][0]);
  const unsigned bs0 = (unsigned)reinterpret_cast<unsigned long long>(&Bs[0][0]);

  v8f acc[2][2] = {};

  stage_tile_async<64>(xb, as0, m0, 0, tid);
  stage_tile_async<128>(wb, bs0, n0, 0, tid);

  for (int k0 = 0; k0 < CC; k0 += 32) {
    const int cur = (k0 >> 5) & 1;
    asm volatile("s_wait_asynccnt 0" ::: "memory");
    __syncthreads();
    if (k0 + 32 < CC) {   // prefetch next tile into the other buffer
      stage_tile_async<64>(xb, as0 + (cur ^ 1) * (64 * 40 * 2), m0, k0 + 32, tid);
      stage_tile_async<128>(wb, bs0 + (cur ^ 1) * (128 * 40 * 2), n0, k0 + 32, tid);
    }
    const __bf16* Ac = &As[cur][0];
    const __bf16* Bc = &Bs[cur][0];
    v16bf a0 = ldfragA(Ac, wm * 32 + lrow, lhi);
    v16bf a1 = ldfragA(Ac, wm * 32 + 16 + lrow, lhi);
    v16bf b0 = ldfragB(Bc, wn * 32 + lrow, lhi);
    v16bf b1 = ldfragB(Bc, wn * 32 + 16 + lrow, lhi);
    acc[0][0] = wmma_bf16(a0, b0, acc[0][0]);
    acc[0][1] = wmma_bf16(a0, b1, acc[0][1]);
    acc[1][0] = wmma_bf16(a1, b0, acc[1][0]);
    acc[1][1] = wmma_bf16(a1, b1, acc[1][1]);
  }

  // Epilogue: RoPE for q/k sections, transpose-scatter for v.
#pragma unroll
  for (int ms = 0; ms < 2; ++ms)
#pragma unroll
    for (int ns = 0; ns < 2; ++ns)
#pragma unroll
      for (int r = 0; r < 8; ++r) {
        int m = m0 + wm * 32 + ms * 16 + r + 8 * lhi;
        int n = n0 + wn * 32 + ns * 16 + lrow;
        float val = acc[ms][ns][r];
        float partner = __shfl_xor(val, 1, 32);  // RoPE pair sits in adjacent lane
        int sec = n >> 10;
        int c = n & (CC - 1);
        int b = m >> 11;
        int t = m & (TT - 1);
        int h = c >> 6;
        int dd = c & (DD - 1);
        if (sec < 2) {  // q or k: RoPE. theta = 10000^(-2p/64)
          int p = dd >> 1;
          float theta = __expf(-9.2103403720f * (float)(2 * p) * (1.0f / 64.0f));
          float ang = (float)t * theta;
          float cs, sn;
          __sincosf(ang, &cs, &sn);
          float o = (dd & 1) ? (val * cs + partner * sn)
                             : (val * cs - partner * sn);
          qk[(((size_t)(sec * BB + b) * HH + h) * TT + t) * DD + dd] = (__bf16)o;
        } else {        // v: store transposed [B][H][D][T]
          vt[(((size_t)b * HH + h) * DD + dd) * TT + t] = (__bf16)val;
        }
      }
}

// ---------------------------------------------------------------------------
// Kernel 2: flash attention. One wave = 16 q rows, block = 128 q rows, one (b,h).
// ---------------------------------------------------------------------------
__global__ __launch_bounds__(256)
void flash_attn(const __bf16* __restrict__ qk, const __bf16* __restrict__ vt,
                __bf16* __restrict__ y) {
  __shared__ __bf16 pbuf[8 * 16 * 40];     // per-wave P relayout buffer
  const int tid = threadIdx.x;
  const int lane = tid & 31, wid = tid >> 5;
  const int lrow = lane & 15, lhi = lane >> 4;
  const int bh = blockIdx.y;               // b*H + h
  const int qt0 = blockIdx.x * 128 + wid * 16;

  const __bf16* Q = qk + (size_t)bh * TT * DD;
  const __bf16* K = qk + (size_t)(BB * HH + bh) * TT * DD;
  const __bf16* V = vt + (size_t)bh * DD * TT;
  __bf16* pb = pbuf + wid * (16 * 40);

  // Q fragments (K=32 each, covering d[0,32) and d[32,64)), kept resident.
  const __bf16* qrow = Q + (size_t)(qt0 + lrow) * DD + 16 * lhi;
  v16bf qf0 = cat8(*(const v8bf*)qrow, *(const v8bf*)(qrow + 8));
  v16bf qf1 = cat8(*(const v8bf*)(qrow + 32), *(const v8bf*)(qrow + 40));

  float mrun[8], lrun[8];
#pragma unroll
  for (int r = 0; r < 8; ++r) { mrun[r] = -3.0e38f; lrun[r] = 0.0f; }
  v8f yacc[4] = {};

  const int qmax = qt0 + 15;
  for (int kt0 = 0; kt0 <= qmax; kt0 += 32) {
    if (kt0 + 32 <= qmax) {  // pull next chunk toward cache (global_prefetch_b8)
      __builtin_prefetch(K + (size_t)(kt0 + 32 + lrow) * DD, 0, 1);
      __builtin_prefetch(V + (size_t)lrow * TT + kt0 + 32, 0, 1);
    }
    // ---- scores: S = Q . K^T for 32 keys (two 16-key column groups) ----
    v8f s0 = {}, s1 = {};
    {
      const __bf16* kr = K + (size_t)(kt0 + lrow) * DD + 16 * lhi;
      v16bf kflo = cat8(*(const v8bf*)kr, *(const v8bf*)(kr + 8));          // d[0,32)
      v16bf kfhi = cat8(*(const v8bf*)(kr + 32), *(const v8bf*)(kr + 40));  // d[32,64)
      s0 = wmma_bf16(qf0, kflo, s0);
      s0 = wmma_bf16(qf1, kfhi, s0);
    }
    {
      const __bf16* kr = K + (size_t)(kt0 + 16 + lrow) * DD + 16 * lhi;
      v16bf kflo = cat8(*(const v8bf*)kr, *(const v8bf*)(kr + 8));
      v16bf kfhi = cat8(*(const v8bf*)(kr + 32), *(const v8bf*)(kr + 40));
      s1 = wmma_bf16(qf0, kflo, s1);
      s1 = wmma_bf16(qf1, kfhi, s1);
    }

    // ---- causal mask + online softmax (rows live across the 16-lane half) ----
    float alpha[8];
#pragma unroll
    for (int r = 0; r < 8; ++r) {
      int tq = qt0 + r + 8 * lhi;
      float a0 = s0[r] * 0.125f;           // 1/sqrt(64)
      float a1 = s1[r] * 0.125f;
      if (kt0 + lrow > tq)      a0 = -3.0e38f;
      if (kt0 + 16 + lrow > tq) a1 = -3.0e38f;
      float rmax = fmaxf(a0, a1);
#pragma unroll
      for (int off = 1; off < 16; off <<= 1)
        rmax = fmaxf(rmax, __shfl_xor(rmax, off, 32));
      float mnew = fmaxf(mrun[r], rmax);
      float al = __expf(mrun[r] - mnew);
      float p0 = __expf(a0 - mnew);
      float p1 = __expf(a1 - mnew);
      float rs = p0 + p1;
#pragma unroll
      for (int off = 1; off < 16; off <<= 1)
        rs += __shfl_xor(rs, off, 32);
      lrun[r] = lrun[r] * al + rs;
      mrun[r] = mnew;
      alpha[r] = al;
      pb[(r + 8 * lhi) * 40 + lrow]      = (__bf16)p0;
      pb[(r + 8 * lhi) * 40 + 16 + lrow] = (__bf16)p1;
    }
#pragma unroll
    for (int g = 0; g < 4; ++g)
#pragma unroll
      for (int r = 0; r < 8; ++r)
        yacc[g][r] = yacc[g][r] * alpha[r];

    // same-wave DS ordering: P stores must land before fragment reads
    asm volatile("s_wait_dscnt 0" ::: "memory");

    const __bf16* pr = pb + lrow * 40 + 16 * lhi;
    v16bf pa = cat8(*(const v8bf*)pr, *(const v8bf*)(pr + 8));

    // ---- y += P . V ; V transposed so B-fragments are contiguous ----
#pragma unroll
    for (int g = 0; g < 4; ++g) {
      const __bf16* vr = V + (size_t)(g * 16 + lrow) * TT + kt0 + 16 * lhi;
      v16bf vf = cat8(*(const v8bf*)vr, *(const v8bf*)(vr + 8));
      yacc[g] = wmma_bf16(pa, vf, yacc[g]);
    }
  }

  // finalize: divide by l, write y as [B][T][C] bf16 for the proj GEMM
  const int b = bh >> 4, h = bh & 15;
#pragma unroll
  for (int g = 0; g < 4; ++g)
#pragma unroll
    for (int r = 0; r < 8; ++r) {
      int tq = qt0 + r + 8 * lhi;
      int ch = h * DD + g * 16 + lrow;
      y[((size_t)b * TT + tq) * CC + ch] = (__bf16)(yacc[g][r] / lrun[r]);
    }
}

// ---------------------------------------------------------------------------
// Kernel 3: out = y @ w_proj^T (M=4096, N=1024, K=1024), f32 out.
// A tile: async-to-LDS loads; B tile: TDM tensor_load_to_lds. Double-buffered.
// ---------------------------------------------------------------------------
__global__ __launch_bounds__(256)
void proj_gemm(const __bf16* __restrict__ yb, const __bf16* __restrict__ wb,
               float* __restrict__ out) {
  __shared__ __bf16 As[2][64 * 40];
  __shared__ __bf16 Bs[2][128 * 40];
  const int tid = threadIdx.x;
  const int lane = tid & 31, wid = tid >> 5;
  const int lrow = lane & 15, lhi = lane >> 4;
  const int wm = wid >> 2, wn = wid & 3;
  const int m0 = blockIdx.y * 64;
  const int n0 = blockIdx.x * 128;
  const unsigned as0 = (unsigned)reinterpret_cast<unsigned long long>(&As[0][0]);
  const unsigned bs0 = (unsigned)reinterpret_cast<unsigned long long>(&Bs[0][0]);

  v8f acc[2][2] = {};

  stage_tile_async<64>(yb, as0, m0, 0, tid);
  if (wid == 0) tdm_load_w_tile(wb, bs0, n0, 0);

  for (int k0 = 0; k0 < CC; k0 += 32) {
    const int cur = (k0 >> 5) & 1;
    asm volatile("s_wait_asynccnt 0" ::: "memory");
    if (wid == 0) __builtin_amdgcn_s_wait_tensorcnt(0);
    __syncthreads();
    if (k0 + 32 < CC) {
      stage_tile_async<64>(yb, as0 + (cur ^ 1) * (64 * 40 * 2), m0, k0 + 32, tid);
      if (wid == 0)
        tdm_load_w_tile(wb, bs0 + (cur ^ 1) * (128 * 40 * 2), n0, k0 + 32);
    }
    const __bf16* Ac = &As[cur][0];
    const __bf16* Bc = &Bs[cur][0];
    v16bf a0 = ldfragA(Ac, wm * 32 + lrow, lhi);
    v16bf a1 = ldfragA(Ac, wm * 32 + 16 + lrow, lhi);
    v16bf b0 = ldfragB(Bc, wn * 32 + lrow, lhi);
    v16bf b1 = ldfragB(Bc, wn * 32 + 16 + lrow, lhi);
    acc[0][0] = wmma_bf16(a0, b0, acc[0][0]);
    acc[0][1] = wmma_bf16(a0, b1, acc[0][1]);
    acc[1][0] = wmma_bf16(a1, b0, acc[1][0]);
    acc[1][1] = wmma_bf16(a1, b1, acc[1][1]);
  }

#pragma unroll
  for (int ms = 0; ms < 2; ++ms)
#pragma unroll
    for (int ns = 0; ns < 2; ++ns)
#pragma unroll
      for (int r = 0; r < 8; ++r) {
        int m = m0 + wm * 32 + ms * 16 + r + 8 * lhi;
        int n = n0 + wn * 32 + ns * 16 + lrow;
        out[(size_t)m * CC + n] = acc[ms][ns][r];
      }
}

extern "C" void kernel_launch(void* const* d_in, const int* in_sizes, int n_in,
                              void* d_out, int out_size, void* d_ws, size_t ws_size,
                              hipStream_t stream) {
  const float* x      = (const float*)d_in[0];  // (B,T,C) f32
  const float* w_qkv  = (const float*)d_in[1];  // (3C,C)  f32
  const float* w_proj = (const float*)d_in[2];  // (C,C)   f32
  float* out = (float*)d_out;                   // (B,T,C) f32

  // workspace (bf16 elems):
  __bf16* qk   = (__bf16*)d_ws;                         //  8,388,608 (q/k roped)
  __bf16* vt   = qk  + (size_t)2 * BB * HH * TT * DD;   //  4,194,304 (v^T)
  __bf16* yws  = vt  + (size_t)BB * HH * DD * TT;       //  4,194,304 (attn out)
  __bf16* xbf  = yws + (size_t)BB * TT * CC;            //  4,194,304
  __bf16* wqb  = xbf + (size_t)BB * TT * CC;            //  3,145,728
  __bf16* wpb  = wqb + (size_t)3 * CC * CC;             //  1,048,576
  // total 25,165,824 bf16 = 48 MB

  const int n4x = BB * TT * CC / 4, n4q = 3 * CC * CC / 4, n4p = CC * CC / 4;
  cvt_f32_bf16<<<dim3(n4x / 256), 256, 0, stream>>>(x, xbf, n4x);
  cvt_f32_bf16<<<dim3(n4q / 256), 256, 0, stream>>>(w_qkv, wqb, n4q);
  cvt_f32_bf16<<<dim3(n4p / 256), 256, 0, stream>>>(w_proj, wpb, n4p);

  qkv_rope_gemm<<<dim3(3 * CC / 128, BB * TT / 64), 256, 0, stream>>>(xbf, wqb, qk, vt);
  flash_attn<<<dim3(TT / 128, BB * HH), 256, 0, stream>>>(qk, vt, yws);
  proj_gemm<<<dim3(CC / 128, BB * TT / 64), 256, 0, stream>>>(yws, wpb, out);
}
